// Decoder_996432412985
// MI455X (gfx1250) — compile-verified, matrix operand order checked by reading, not run on hardware
//
#include <hip/hip_runtime.h>

// ---------------------------------------------------------------------------
// Types for CDNA5 WMMA (gfx1250, wave32)
// ---------------------------------------------------------------------------
typedef __bf16 bf16_t;
typedef __attribute__((ext_vector_type(16))) __bf16 v16bf;
typedef __attribute__((ext_vector_type(8)))  float  v8f;

// ---------------------------------------------------------------------------
// Problem dimensions
// ---------------------------------------------------------------------------
#define NB      64      // batch
#define FMAPC   512     // feature map channels
#define ENC     256     // embedding dim
#define DEC     512     // LSTM hidden dim
#define NATTN   256     // attention dim
#define TSTEPS  31      // MAX_LEN + 1
#define HHH     8
#define WWW     32
#define HW      256     // H*W
#define NOUT    98      // NUM_CLASSES + 2
#define NPAD    112     // NOUT padded to multiple of 16
#define GN      2048    // 4*DEC gate width
#define K0CAT   768     // ENC + DEC
#define K1CAT   1024    // DEC + DEC

// ---------------------------------------------------------------------------
// Workspace layout (bytes, all 256B aligned)
// ---------------------------------------------------------------------------
#define OFF_WCAT0  ((size_t)0)                        // bf16 [2048][768]
#define OFF_WCAT1  (OFF_WCAT0 + 3145728)              // bf16 [2048][1024]
#define OFF_ELSTM  (OFF_WCAT1 + 4194304)              // bf16 [256][512]
#define OFF_FCW    (OFF_ELSTM + 262144)               // bf16 [112][1024]
#define OFF_WTAPS  (OFF_FCW   + 229376)               // bf16 [9][256][512]
#define OFF_FMAPT  (OFF_WTAPS + 2359296)              // bf16 [64*8*32][512] (channels-last)
#define OFF_B0     (OFF_FMAPT + 16777216)             // f32  [2048] bih0+bhh0
#define OFF_B1     (OFF_B0    + 8192)                 // f32  [2048]
#define OFF_FCB    (OFF_B1    + 8192)                 // f32  [112]
#define OFF_EFMAP  (OFF_FCB   + 512)                  // f32  [64][256][256] conv output
#define OFF_GATES  (OFF_EFMAP + 16777216)             // f32  [64][2048]
#define OFF_EH     (OFF_GATES + 524288)               // f32  [64][256]
#define OFF_C0S    (OFF_EH    + 65536)                // f32  [64][512]
#define OFF_C1S    (OFF_C0S   + 131072)               // f32  [64][512]
#define OFF_XH0    (OFF_C1S   + 131072)               // bf16 [64][768]  = [x | h0]
#define OFF_XH1    (OFF_XH0   + 98304)                // bf16 [64][1024] = [h0 | h1]
#define OFF_FCIN   (OFF_XH1   + 131072)               // bf16 [64][1024] = [h1 | glimpse]

// ---------------------------------------------------------------------------
// Tile loaders matching the 16-bit 16x32 A-matrix VGPR layout (ISA 7.12.2):
//   lanes 0-15  hold M=lane,    K in {0..7} U {16..23}
//   lanes 16-31 hold M=lane-16, K in {8..15} U {24..31}
// Each lane's 16 bf16 values = two contiguous 16-byte runs.
// B (K x N) operand uses the symmetric layout with N striped across lanes,
// which for row-major W[N][K] means the loader is identical with W as "A".
// ---------------------------------------------------------------------------
__device__ __forceinline__ v16bf ldrow(const bf16_t* rowp, int k0, int lane) {
  int kh = (lane >> 4) << 3;
  union { float4 f[2]; v16bf v; } u;
  u.f[0] = *reinterpret_cast<const float4*>(rowp + k0 + kh);
  u.f[1] = *reinterpret_cast<const float4*>(rowp + k0 + 16 + kh);
  return u.v;
}

__device__ __forceinline__ v16bf ldtile(const bf16_t* base, int ld, int row0,
                                        int k0, int lane) {
  return ldrow(base + (size_t)(row0 + (lane & 15)) * ld, k0, lane);
}

__device__ __forceinline__ v8f zero8() {
  v8f z;
#pragma unroll
  for (int i = 0; i < 8; ++i) z[i] = 0.f;
  return z;
}

// ---------------------------------------------------------------------------
// Generic wave-level GEMM:  C[M=64, N] = A[64, K](bf16) * W[N, K]^T(bf16) + bias
// One wave per block; each wave owns one 16-wide N slice and all four M tiles
// (B tile loaded once, reused 4x). f32 accumulation via v_wmma_f32_16x16x32_bf16.
// Used for the latency-critical recurrent GEMMs: data is L1/L2 resident, so no
// LDS staging (barriers would only lengthen the 31-step serial chain).
// ---------------------------------------------------------------------------
__global__ __launch_bounds__(32)
void wmma_gemm(const bf16_t* __restrict__ A, int lda,
               const bf16_t* __restrict__ W, int ldw,
               const float* __restrict__ bias,
               float* __restrict__ C, int ldc, int nLimit, int K) {
  int lane = threadIdx.x;
  int nb = blockIdx.x * 16;
  int mg = blockIdx.y * 64;

  v8f acc[4];
#pragma unroll
  for (int i = 0; i < 4; ++i) acc[i] = zero8();

  for (int k0 = 0; k0 < K; k0 += 32) {
    v16bf bt = ldtile(W, ldw, nb, k0, lane);
#pragma unroll
    for (int i = 0; i < 4; ++i) {
      v16bf at = ldtile(A, lda, mg + i * 16, k0, lane);
      acc[i] = __builtin_amdgcn_wmma_f32_16x16x32_bf16(
          false, at, false, bt, (short)0, acc[i], false, false);
    }
  }

  int n = nb + (lane & 15);
  float bv = bias ? bias[n] : 0.f;
  if (n < nLimit) {
#pragma unroll
    for (int i = 0; i < 4; ++i) {
#pragma unroll
      for (int v = 0; v < 8; ++v) {
        int m = mg + i * 16 + ((lane >> 4) << 3) + v;
        C[(size_t)m * ldc + n] = acc[i][v] + bv;
      }
    }
  }
}

// ---------------------------------------------------------------------------
// 3x3 SAME conv on the (constant) feature map as 9 shifted WMMA GEMMs.
// 8-wave workgroup computes a 64-row x 128-channel output slab. The activation
// slab (4 halo rows x 34 halo cols x 32-channel chunk, zero-padded borders) is
// shared by all 8 waves and double-buffered in LDS via ASYNC loads
// (global_load_async_to_lds_b128, ASYNCcnt), so chunk k+1 streams in while the
// waves run 9 taps x 4 tiles of WMMA on chunk k out of LDS.
// Inner LDS stride padded 32->40 bf16 so the 16 lanes' 80B-strided
// ds_load_b128 land in disjoint bank groups (stride 20 dwords, gcd(20,64)=4).
// ---------------------------------------------------------------------------
__global__ __launch_bounds__(256)
void conv3x3_wmma(const bf16_t* __restrict__ fmapt,
                  const bf16_t* __restrict__ wtaps,
                  const float* __restrict__ bias,
                  float* __restrict__ efmap) {
  __shared__ __align__(16) bf16_t slab[2][4][34][40];

  int tid  = threadIdx.x;
  int lane = tid & 31;
  int wv   = tid >> 5;
  int nb   = blockIdx.x * 128 + wv * 16;   // attn channel base for this wave
  int mg   = blockIdx.y * 64;              // row base for this block
  int b    = mg >> 8;
  int h0   = (mg >> 5) & 7;                // block covers rows h0, h0+1 (all w)

  v8f acc[4];
#pragma unroll
  for (int i = 0; i < 4; ++i) acc[i] = zero8();

  // Stage one 32-channel chunk of the (zero-padded) activation halo into LDS.
  // 4*34 positions x 64B = 544 x 16B async transfers spread over 256 threads.
  auto stage = [&](int sel, int k0) {
    for (int q = tid; q < 4 * 34 * 4; q += 256) {
      int p = q >> 2, part = q & 3;
      int sy = p / 34, sx = p % 34;
      int y = h0 - 1 + sy, x = sx - 1;
      bf16_t* lp = &slab[sel][sy][sx][part * 8];
      if (y >= 0 && y < HHH && x >= 0 && x < WWW) {
        const bf16_t* gp =
            fmapt + (size_t)((b * HHH + y) * WWW + x) * FMAPC + k0 + part * 8;
        unsigned loff = (unsigned)(uintptr_t)lp;   // low 32 bits = LDS offset
        asm volatile("global_load_async_to_lds_b128 %0, %1, off"
                     :: "v"(loff), "v"(gp) : "memory");
      } else {
        *reinterpret_cast<float4*>(lp) = make_float4(0.f, 0.f, 0.f, 0.f);
      }
    }
  };

  stage(0, 0);
  for (int kc = 0; kc < FMAPC / 32; ++kc) {
    int sel = kc & 1;
    asm volatile("s_wait_asynccnt 0x0" ::: "memory");
    __syncthreads();   // slab[sel] ready for all waves; prev reads of slab[sel^1] done
    if (kc + 1 < FMAPC / 32) stage(sel ^ 1, (kc + 1) * 32);

    int k0 = kc * 32;
    for (int tap = 0; tap < 9; ++tap) {
      int dy = tap / 3 - 1, dx = tap % 3 - 1;
      v16bf bt =
          ldtile(wtaps + (size_t)tap * NATTN * FMAPC, FMAPC, nb, k0, lane);
#pragma unroll
      for (int i = 0; i < 4; ++i) {
        int sy = (i >> 1) + dy + 1;                       // 0..3
        int sx = (i & 1) * 16 + (lane & 15) + dx + 1;     // 0..33
        v16bf at = ldrow(&slab[sel][sy][sx][0], 0, lane);
        acc[i] = __builtin_amdgcn_wmma_f32_16x16x32_bf16(
            false, at, false, bt, (short)0, acc[i], false, false);
      }
    }
  }

  int a = nb + (lane & 15);
  float bv = bias[a];
#pragma unroll
  for (int i = 0; i < 4; ++i) {
#pragma unroll
    for (int v = 0; v < 8; ++v) {
      int m = mg + i * 16 + ((lane >> 4) << 3) + v;
      int bb = m >> 8, h = (m >> 5) & 7, w = m & 31;
      efmap[((size_t)bb * NATTN + a) * HW + h * WWW + w] = acc[i][v] + bv;
    }
  }
}

// ---------------------------------------------------------------------------
// Elementwise LSTM cell: gates [64][2048] (i,f,g,o), updates c in place,
// writes new h (bf16) into the two consumer buffers.
// ---------------------------------------------------------------------------
__device__ __forceinline__ float sigf(float x) { return 1.f / (1.f + __expf(-x)); }

__global__ __launch_bounds__(256)
void lstm_cell(const float* __restrict__ gates, float* __restrict__ c,
               bf16_t* __restrict__ hdst1, int ld1,
               bf16_t* __restrict__ hdst2, int ld2) {
  int idx = blockIdx.x * 256 + threadIdx.x;       // 64*512
  int b = idx >> 9, j = idx & 511;
  const float* g = gates + (size_t)b * GN;
  float gi = sigf(g[j]);
  float gf = sigf(g[512 + j]);
  float gg = tanhf(g[1024 + j]);
  float go = sigf(g[1536 + j]);
  float cn = gf * c[idx] + gi * gg;
  c[idx] = cn;
  float h = go * tanhf(cn);
  hdst1[(size_t)b * ld1 + j] = (bf16_t)h;
  hdst2[(size_t)b * ld2 + j] = (bf16_t)h;
}

// ---------------------------------------------------------------------------
// Embedding lookup (teacher forcing: label[0]=BOS 0, label[t]=target[:,t-1]),
// written as bf16 into the first ENC columns of xh0.
// ---------------------------------------------------------------------------
__global__ __launch_bounds__(256)
void embed_step(const float* __restrict__ emb, const int* __restrict__ target,
                bf16_t* __restrict__ xh0, int t) {
  int idx = blockIdx.x * 256 + threadIdx.x;       // 64*256
  int b = idx >> 8, e = idx & 255;
  int lab = (t == 0) ? 0 : target[b * (TSTEPS - 1) + (t - 1)];
  xh0[(size_t)b * K0CAT + e] = (bf16_t)emb[(size_t)lab * ENC + e];
}

// ---------------------------------------------------------------------------
// Attention: score -> softmax over HW -> mask out, glimpse out (+bf16 into fcin)
// One block per batch element, one thread per spatial position.
// ---------------------------------------------------------------------------
__global__ __launch_bounds__(256)
void attention_step(const float* __restrict__ eh, const float* __restrict__ efmap,
                    const float* __restrict__ fmap,
                    const float* __restrict__ a_w, const float* __restrict__ a_b,
                    float* __restrict__ out_masks, float* __restrict__ out_glimpse,
                    bf16_t* __restrict__ fcin, int t) {
  __shared__ float sh[HW];
  __shared__ float red[HW];
  int b = blockIdx.x, tid = threadIdx.x;

  sh[tid] = eh[b * NATTN + tid];
  __syncthreads();

  // score[tid] = sum_a a_w[a] * tanh(e_h[a] + e_fmap[b,a,tid]) + a_b
  float s = 0.f;
  const float* ef = efmap + (size_t)b * NATTN * HW + tid;
  for (int a = 0; a < NATTN; ++a)
    s += a_w[a] * tanhf(sh[a] + ef[(size_t)a * HW]);
  s += a_b[0];

  // softmax over HW
  red[tid] = s; __syncthreads();
  for (int st = 128; st > 0; st >>= 1) {
    if (tid < st) red[tid] = fmaxf(red[tid], red[tid + st]);
    __syncthreads();
  }
  float mx = red[0]; __syncthreads();
  float e = __expf(s - mx);
  red[tid] = e; __syncthreads();
  for (int st = 128; st > 0; st >>= 1) {
    if (tid < st) red[tid] += red[tid + st];
    __syncthreads();
  }
  float mask = e / red[0];
  __syncthreads();
  sh[tid] = mask;
  out_masks[((size_t)b * TSTEPS + t) * HW + tid] = mask;
  __syncthreads();

  // glimpse[c] = sum_hw fmap[b,c,hw] * mask[hw]
  for (int c = tid; c < FMAPC; c += 256) {
    const float* fp = fmap + ((size_t)b * FMAPC + c) * HW;
    float g = 0.f;
    for (int hw = 0; hw < HW; ++hw) g += fp[hw] * sh[hw];
    out_glimpse[((size_t)b * TSTEPS + t) * FMAPC + c] = g;
    fcin[(size_t)b * K1CAT + DEC + c] = (bf16_t)g;
  }
}

// ---------------------------------------------------------------------------
// One-time packing kernels (weights -> bf16 WMMA layouts, fmap -> channels-last)
// ---------------------------------------------------------------------------
__global__ __launch_bounds__(256)
void pack_wcat0(const float* __restrict__ Wih, const float* __restrict__ Whh,
                bf16_t* __restrict__ dst) {
  int i = blockIdx.x * 256 + threadIdx.x;
  if (i >= GN * K0CAT) return;
  int n = i / K0CAT, k = i % K0CAT;
  float v = (k < ENC) ? Wih[(size_t)n * ENC + k] : Whh[(size_t)n * DEC + (k - ENC)];
  dst[i] = (bf16_t)v;
}

__global__ __launch_bounds__(256)
void pack_wcat1(const float* __restrict__ Wih, const float* __restrict__ Whh,
                bf16_t* __restrict__ dst) {
  int i = blockIdx.x * 256 + threadIdx.x;
  if (i >= GN * K1CAT) return;
  int n = i / K1CAT, k = i % K1CAT;
  float v = (k < DEC) ? Wih[(size_t)n * DEC + k] : Whh[(size_t)n * DEC + (k - DEC)];
  dst[i] = (bf16_t)v;
}

__global__ __launch_bounds__(256)
void pack_f2bf(const float* __restrict__ src, bf16_t* __restrict__ dst, int n) {
  int i = blockIdx.x * 256 + threadIdx.x;
  if (i < n) dst[i] = (bf16_t)src[i];
}

__global__ __launch_bounds__(256)
void pack_fc(const float* __restrict__ fc_w, const float* __restrict__ fc_b,
             bf16_t* __restrict__ dst, float* __restrict__ fcb) {
  int i = blockIdx.x * 256 + threadIdx.x;
  if (i >= NPAD * K1CAT) return;
  int n = i / K1CAT, k = i % K1CAT;
  dst[i] = (bf16_t)((n < NOUT) ? fc_w[(size_t)n * K1CAT + k] : 0.f);
  if (k == 0) fcb[n] = (n < NOUT) ? fc_b[n] : 0.f;
}

__global__ __launch_bounds__(256)
void pack_bias(const float* __restrict__ bih0, const float* __restrict__ bhh0,
               const float* __restrict__ bih1, const float* __restrict__ bhh1,
               float* __restrict__ b0, float* __restrict__ b1) {
  int i = blockIdx.x * 256 + threadIdx.x;
  if (i >= GN) return;
  b0[i] = bih0[i] + bhh0[i];
  b1[i] = bih1[i] + bhh1[i];
}

__global__ __launch_bounds__(256)
void pack_wtaps(const float* __restrict__ w, bf16_t* __restrict__ dst) {
  int i = blockIdx.x * 256 + threadIdx.x;          // 9*256*512
  if (i >= 9 * NATTN * FMAPC) return;
  int tap = i / (NATTN * FMAPC);
  int r = i % (NATTN * FMAPC);
  int a = r / FMAPC, c = r % FMAPC;
  dst[i] = (bf16_t)w[((size_t)a * FMAPC + c) * 9 + tap];
}

__global__ __launch_bounds__(256)
void pack_fmapt(const float* __restrict__ fmap, bf16_t* __restrict__ dst) {
  int i = blockIdx.x * 256 + threadIdx.x;          // 64*256*512
  if (i >= NB * HW * FMAPC) return;
  int c = i % FMAPC;
  int r = i / FMAPC;                               // (b*8+h)*32+w
  int b = r >> 8, hw = r & 255;
  dst[i] = (bf16_t)fmap[((size_t)b * FMAPC + c) * HW + hw];
}

__global__ __launch_bounds__(256)
void init_states(const float* __restrict__ h0, const float* __restrict__ c0,
                 float* __restrict__ c0s, float* __restrict__ c1s,
                 bf16_t* __restrict__ xh0, bf16_t* __restrict__ xh1) {
  int i = blockIdx.x * 256 + threadIdx.x;          // 64*512
  if (i >= NB * DEC) return;
  int b = i >> 9, d = i & 511;
  c0s[i] = c0[i];
  c1s[i] = c0[NB * DEC + i];
  xh0[(size_t)b * K0CAT + ENC + d] = (bf16_t)h0[i];
  xh1[(size_t)b * K1CAT + DEC + d] = (bf16_t)h0[NB * DEC + i];
}

// ---------------------------------------------------------------------------
// Host launcher
// ---------------------------------------------------------------------------
extern "C" void kernel_launch(void* const* d_in, const int* in_sizes, int n_in,
                              void* d_out, int out_size, void* d_ws, size_t ws_size,
                              hipStream_t stream) {
  (void)in_sizes; (void)n_in; (void)out_size; (void)ws_size;

  const float* fmap    = (const float*)d_in[0];
  const float* h0      = (const float*)d_in[1];
  const float* c0      = (const float*)d_in[2];
  const int*   target  = (const int*)  d_in[3];
  /* d_in[4] = length, unused in forward math */
  const float* emb     = (const float*)d_in[5];
  const float* Wih0    = (const float*)d_in[6];
  const float* Whh0    = (const float*)d_in[7];
  const float* bih0    = (const float*)d_in[8];
  const float* bhh0    = (const float*)d_in[9];
  const float* Wih1    = (const float*)d_in[10];
  const float* Whh1    = (const float*)d_in[11];
  const float* bih1    = (const float*)d_in[12];
  const float* bhh1    = (const float*)d_in[13];
  const float* e_lstm_w = (const float*)d_in[14];
  const float* e_lstm_b = (const float*)d_in[15];
  const float* e_fmap_w = (const float*)d_in[16];
  const float* e_fmap_b = (const float*)d_in[17];
  const float* a_w     = (const float*)d_in[18];
  const float* a_b     = (const float*)d_in[19];
  const float* fc_w    = (const float*)d_in[20];
  const float* fc_b    = (const float*)d_in[21];

  char* ws = (char*)d_ws;
  bf16_t* WCAT0 = (bf16_t*)(ws + OFF_WCAT0);
  bf16_t* WCAT1 = (bf16_t*)(ws + OFF_WCAT1);
  bf16_t* ELSTM = (bf16_t*)(ws + OFF_ELSTM);
  bf16_t* FCW   = (bf16_t*)(ws + OFF_FCW);
  bf16_t* WTAPS = (bf16_t*)(ws + OFF_WTAPS);
  bf16_t* FMAPT = (bf16_t*)(ws + OFF_FMAPT);
  float*  B0    = (float*) (ws + OFF_B0);
  float*  B1    = (float*) (ws + OFF_B1);
  float*  FCB   = (float*) (ws + OFF_FCB);
  float*  EFMAP = (float*) (ws + OFF_EFMAP);
  float*  GATES = (float*) (ws + OFF_GATES);
  float*  EH    = (float*) (ws + OFF_EH);
  float*  C0S   = (float*) (ws + OFF_C0S);
  float*  C1S   = (float*) (ws + OFF_C1S);
  bf16_t* XH0   = (bf16_t*)(ws + OFF_XH0);
  bf16_t* XH1   = (bf16_t*)(ws + OFF_XH1);
  bf16_t* FCIN  = (bf16_t*)(ws + OFF_FCIN);

  float* out        = (float*)d_out;
  float* out_logits = out;                                   // [64][31][98]
  float* out_masks  = out + (size_t)NB * TSTEPS * NOUT;      // [64][31][256]
  float* out_glimps = out_masks + (size_t)NB * TSTEPS * HW;  // [64][31][512]

  // ---- one-time packing ----
  pack_wcat0 <<<(GN * K0CAT + 255) / 256, 256, 0, stream>>>(Wih0, Whh0, WCAT0);
  pack_wcat1 <<<(GN * K1CAT + 255) / 256, 256, 0, stream>>>(Wih1, Whh1, WCAT1);
  pack_f2bf  <<<(NATTN * DEC + 255) / 256, 256, 0, stream>>>(e_lstm_w, ELSTM, NATTN * DEC);
  pack_fc    <<<(NPAD * K1CAT + 255) / 256, 256, 0, stream>>>(fc_w, fc_b, FCW, FCB);
  pack_bias  <<<(GN + 255) / 256, 256, 0, stream>>>(bih0, bhh0, bih1, bhh1, B0, B1);
  pack_wtaps <<<(9 * NATTN * FMAPC + 255) / 256, 256, 0, stream>>>(e_fmap_w, WTAPS);
  pack_fmapt <<<(NB * HW * FMAPC + 255) / 256, 256, 0, stream>>>(fmap, FMAPT);
  init_states<<<(NB * DEC + 255) / 256, 256, 0, stream>>>(h0, c0, C0S, C1S, XH0, XH1);

  // ---- one-time 3x3 conv on feature map: e_fmap = conv(fmap) + bias ----
  // 8 waves/block, 64 rows x 128 attn channels, async double-buffered LDS.
  conv3x3_wmma<<<dim3(NATTN / 128, (NB * HW) / 64), 256, 0, stream>>>(
      FMAPT, WTAPS, e_fmap_b, EFMAP);

  // ---- 31 sequential decoder steps ----
  for (int t = 0; t < TSTEPS; ++t) {
    embed_step<<<NB, 256, 0, stream>>>(emb, target, XH0, t);

    // LSTM layer 0: gates = [x|h0] @ Wcat0^T + (bih0+bhh0)
    wmma_gemm<<<dim3(GN / 16, 1), 32, 0, stream>>>(
        XH0, K0CAT, WCAT0, K0CAT, B0, GATES, GN, GN, K0CAT);
    lstm_cell<<<(NB * DEC) / 256, 256, 0, stream>>>(
        GATES, C0S, XH0 + ENC, K0CAT, XH1, K1CAT);

    // LSTM layer 1: gates = [h0|h1] @ Wcat1^T + (bih1+bhh1)
    wmma_gemm<<<dim3(GN / 16, 1), 32, 0, stream>>>(
        XH1, K1CAT, WCAT1, K1CAT, B1, GATES, GN, GN, K1CAT);
    lstm_cell<<<(NB * DEC) / 256, 256, 0, stream>>>(
        GATES, C1S, XH1 + DEC, K1CAT, FCIN, K1CAT);

    // e_h = h1 @ e_lstm_w^T + e_lstm_b   (h1 lives in FCIN[:, :512])
    wmma_gemm<<<dim3(NATTN / 16, 1), 32, 0, stream>>>(
        FCIN, K1CAT, ELSTM, DEC, e_lstm_b, EH, NATTN, NATTN, DEC);

    // attention: tanh + score + softmax + glimpse (writes masks/glimpses/fcin)
    attention_step<<<NB, 256, 0, stream>>>(
        EH, EFMAP, fmap, a_w, a_b, out_masks, out_glimps, FCIN, t);

    // logits = [h1|glimpse] @ fc_w^T + fc_b  (N padded to 112, store 98)
    wmma_gemm<<<dim3(NPAD / 16, 1), 32, 0, stream>>>(
        FCIN, K1CAT, FCW, K1CAT, FCB, out_logits + t * NOUT,
        TSTEPS * NOUT, NOUT, K1CAT);
  }
}
